// TextGCNConv_37434934952834
// MI455X (gfx1250) — matrix operand order repaired and checked by reading, not compiled, not run on hardware
//
#include <hip/hip_runtime.h>

typedef __attribute__((ext_vector_type(16))) _Float16 v16h;
typedef __attribute__((ext_vector_type(8)))  _Float16 v8h;
typedef __attribute__((ext_vector_type(8)))  float    v8f;
typedef __attribute__((ext_vector_type(4)))  float    v4f;

#define B_ 32
#define N_ 128
#define D_ 128
#define E_ 64

// ---------------------------------------------------------------------------
// Kernel 1: xlb[b,n,k] = sum_d x[b,n,d] * Wx[k,d] + bx[k] + be[k]
// One workgroup per batch b (8 waves). Wave w owns output columns [16w,16w+16).
// x[b] staged in LDS as f16; Wx B-fragments built from global (L2-resident).
// ---------------------------------------------------------------------------
#define XS_STRIDE 136   // halfs per row: 272B -> 16B aligned rows, conflict-free TR reads

__global__ __launch_bounds__(256) void xl_kernel(
    const float* __restrict__ x, const float* __restrict__ Wx,
    const float* __restrict__ bx, const float* __restrict__ be,
    float* __restrict__ xlb)
{
  __shared__ _Float16 xs[N_ * XS_STRIDE];
  const int b   = blockIdx.x;
  const int tid = threadIdx.x;

  // stage x[b] (128x128 f32) -> f16 LDS
  const float* xb = x + (size_t)b * N_ * D_;
  for (int idx = tid; idx < N_ * D_ / 4; idx += 256) {
    v4f v = *(const v4f*)(xb + idx * 4);
    int row = (idx * 4) / D_;
    int col = (idx * 4) % D_;
    _Float16* p = &xs[row * XS_STRIDE + col];
    p[0] = (_Float16)v[0]; p[1] = (_Float16)v[1];
    p[2] = (_Float16)v[2]; p[3] = (_Float16)v[3];
  }
  __syncthreads();

  const int wave = tid >> 5;
  const int lane = tid & 31;
  const int hl   = lane >> 4;     // which half-wave
  const int lr   = lane & 15;
  const int k0   = wave * 16;     // output column tile

  // B fragments (Wx transposed on the fly): Bc[t] = Wx[k0+lr][32c + 8*hl + t + (t & 8)]
  v16h Bf[4];
  const float* wrow = Wx + (size_t)(k0 + lr) * D_;
  #pragma unroll
  for (int c = 0; c < 4; ++c) {
    const float* p = wrow + 32 * c + 8 * hl;
    v16h bb;
    #pragma unroll
    for (int t = 0; t < 16; ++t) bb[t] = (_Float16)p[t + (t & 8)];
    Bf[c] = bb;
  }

  const float biask = bx[k0 + lr] + be[k0 + lr];

  for (int nt = 0; nt < 8; ++nt) {
    v8f acc = {};
    #pragma unroll
    for (int c = 0; c < 4; ++c) {
      // A fragment: row M = nt*16+lr; K halves split across half-waves
      const _Float16* pr = &xs[(nt * 16 + lr) * XS_STRIDE + 32 * c + 8 * hl];
      v8h lo = *(const v8h*)pr;
      v8h hi = *(const v8h*)(pr + 16);
      v16h A;
      #pragma unroll
      for (int t = 0; t < 8; ++t) { A[t] = lo[t]; A[t + 8] = hi[t]; }
      acc = __builtin_amdgcn_wmma_f32_16x16x32_f16(false, A, false, Bf[c],
                                                   (short)0, acc, false, false);
    }
    #pragma unroll
    for (int r = 0; r < 8; ++r) {
      int m = r + 8 * hl;     // C/D layout: VGPR r = rows r (lanes 0-15) / r+8 (16-31)
      xlb[((size_t)b * N_ + nt * 16 + m) * D_ + k0 + lr] = acc[r] + biask;
    }
  }
}

// ---------------------------------------------------------------------------
// Kernel 2: rdi[b,i] = deg_row^-1/2 ; rdj[b,j] = deg_col^-1/2 (0 if degree 0)
// ---------------------------------------------------------------------------
__global__ __launch_bounds__(128) void degree_kernel(
    const int* __restrict__ adj, float* __restrict__ rdi, float* __restrict__ rdj)
{
  const int b = blockIdx.x;
  const int t = threadIdx.x;
  const int* ab = adj + (size_t)b * N_ * N_;
  int dr = 0, dc = 0;
  for (int j = 0; j < N_; ++j) {
    dr += (ab[t * N_ + j] != 0);
    dc += (ab[j * N_ + t] != 0);
  }
  rdi[b * N_ + t] = dr > 0 ? rsqrtf((float)dr) : 0.f;
  rdj[b * N_ + t] = dc > 0 ? rsqrtf((float)dc) : 0.f;
}

// ---------------------------------------------------------------------------
// Kernel 3 (hot): one WG per (b,i).
//   el = e[b,i] @ We^T  via WMMA (K=64 -> 2 chained wmma per 16x16 tile)
//   fused: + xlb[b,j,:], relu, * (adj * dj^-1/2), reduce over j, * di^-1/2
// ---------------------------------------------------------------------------
#define ES_STRIDE 72    // halfs per row: 144B -> 16B aligned, conflict-free

__global__ __launch_bounds__(256) void gcn_main_kernel(
    const int* __restrict__ adj, const float* __restrict__ e,
    const float* __restrict__ We, const float* __restrict__ xlb,
    const float* __restrict__ rdi, const float* __restrict__ rdj,
    float* __restrict__ out)
{
  __shared__ _Float16 es[N_ * ES_STRIDE];
  __shared__ float    s[N_];
  const int bi  = blockIdx.x;       // b*128 + i
  const int b   = bi >> 7;
  const int tid = threadIdx.x;

  // stage e[b,i] (128x64 f32 = 32KB) -> f16 LDS (coalesced 16B global loads)
  const float* ebi = e + (size_t)bi * N_ * E_;
  for (int idx = tid; idx < N_ * E_ / 4; idx += 256) {
    v4f v = *(const v4f*)(ebi + idx * 4);
    int row = idx >> 4;             // (idx*4)/64
    int col = (idx & 15) * 4;
    _Float16* p = &es[row * ES_STRIDE + col];
    p[0] = (_Float16)v[0]; p[1] = (_Float16)v[1];
    p[2] = (_Float16)v[2]; p[3] = (_Float16)v[3];
  }
  // per-row scale s[j] = adj[b,i,j] ? dj^-1/2 : 0   (mask+norm folded)
  if (tid < N_) {
    int a = adj[(size_t)bi * N_ + tid];
    s[tid] = a ? rdj[b * N_ + tid] : 0.f;
  }
  __syncthreads();

  const int wave = tid >> 5;
  const int lane = tid & 31;
  const int hl   = lane >> 4;
  const int lr   = lane & 15;
  const int d0   = wave * 16;       // this wave's 16 output columns

  // B fragments (We transposed on the fly): Bc[t] = We[d0+lr][32c + 8*hl + t + (t & 8)]
  v16h Bf[2];
  const float* wrow = We + (size_t)(d0 + lr) * E_;
  #pragma unroll
  for (int c = 0; c < 2; ++c) {
    const float* p = wrow + 32 * c + 8 * hl;
    v16h bb;
    #pragma unroll
    for (int t = 0; t < 16; ++t) bb[t] = (_Float16)p[t + (t & 8)];
    Bf[c] = bb;
  }

  float sum = 0.f;
  const float* xlbase = xlb + (size_t)b * N_ * D_ + d0 + lr;

  for (int jt = 0; jt < 8; ++jt) {
    v8f acc = {};
    #pragma unroll
    for (int c = 0; c < 2; ++c) {
      const _Float16* pr = &es[(jt * 16 + lr) * ES_STRIDE + 32 * c + 8 * hl];
      v8h lo = *(const v8h*)pr;
      v8h hi = *(const v8h*)(pr + 16);
      v16h A;
      #pragma unroll
      for (int t = 0; t < 8; ++t) { A[t] = lo[t]; A[t + 8] = hi[t]; }
      acc = __builtin_amdgcn_wmma_f32_16x16x32_f16(false, A, false, Bf[c],
                                                   (short)0, acc, false, false);
    }
    // fused epilogue + reduction over j within this tile
    #pragma unroll
    for (int r = 0; r < 8; ++r) {
      int m = r + 8 * hl;
      int j = jt * 16 + m;
      float val = acc[r] + xlbase[(size_t)j * D_];   // el + xl (+biases)
      val = fmaxf(val, 0.f);                         // relu
      sum += val * s[j];                             // mask * dj^-1/2, sum_j
    }
  }
  // lanes L and L+16 hold partial sums of the same column (rows 0-7 vs 8-15)
  sum += __shfl_xor(sum, 16, 32);
  if (lane < 16)
    out[(size_t)bi * D_ + d0 + lr] = rdi[bi] * sum;  // * di^-1/2
}

// ---------------------------------------------------------------------------
extern "C" void kernel_launch(void* const* d_in, const int* in_sizes, int n_in,
                              void* d_out, int out_size, void* d_ws, size_t ws_size,
                              hipStream_t stream) {
  const float* x   = (const float*)d_in[0];
  const int*   adj = (const int*)  d_in[1];
  const float* e   = (const float*)d_in[2];
  const float* Wx  = (const float*)d_in[3];
  const float* bx  = (const float*)d_in[4];
  const float* We  = (const float*)d_in[5];
  const float* be  = (const float*)d_in[6];
  float* out = (float*)d_out;

  float* xlb = (float*)d_ws;          // [B,N,D]  x@Wx^T + bx + be
  float* rdi = xlb + B_ * N_ * D_;    // [B,N]    row-degree^-1/2
  float* rdj = rdi + B_ * N_;         // [B,N]    col-degree^-1/2

  xl_kernel<<<B_, 256, 0, stream>>>(x, Wx, bx, be, xlb);
  degree_kernel<<<B_, 128, 0, stream>>>(adj, rdi, rdj);
  gcn_main_kernel<<<B_ * N_, 256, 0, stream>>>(adj, e, We, xlb, rdi, rdj, out);
}
